// Doublel_kan_10316511445865
// MI455X (gfx1250) — matrix operand order repaired and checked by reading, not compiled, not run on hardware
//
#include <hip/hip_runtime.h>

// ---------------------------------------------------------------------------
// KAN double layer for MI455X (gfx1250, wave32, WMMA).
// Each layer collapses to one GEMM: C[8192,1024] = A_aug[8192,9216] x W_aug^T
// where per input feature we store 8 cubic-B-spline bases + the raw x value
// (augmented K = 1024*9 = 9216), and W_aug packs spline_w (8) + base_w (1).
// f16 operands, f32 accumulation via v_wmma_f32_16x16x32_f16.
// All device-memory accesses are cast to address_space(1) so the backend
// emits global_load/global_store/global_prefetch instead of flat ops
// (flat ops needlessly couple LOADcnt with DScnt in the inner loop).
// ---------------------------------------------------------------------------

typedef _Float16 half8   __attribute__((ext_vector_type(8)));
typedef _Float16 half16  __attribute__((ext_vector_type(16)));
typedef float    floatx8 __attribute__((ext_vector_type(8)));

#define GLOBAL(T) T __attribute__((address_space(1)))

#define N_TOK 8192
#define CH    1024
#define KC    8           // spline coefficients per feature
#define KAUG  9           // 8 bases + raw x
#define KDIM  (CH * KAUG) // 9216, multiple of 32

#define A_BYTES ((size_t)N_TOK * KDIM * 2)   // 150,994,944
#define W_BYTES ((size_t)CH    * KDIM * 2)   //  18,874,368

// ---------------------------------------------------------------------------
// Pack W_aug[o][i*9+k] = spline_w[o][i][k] (k<8) ; base_w[o][i] (k==8), as f16
// ---------------------------------------------------------------------------
__global__ __launch_bounds__(256) void pack_weights(
    const float* __restrict__ base_w,
    const float* __restrict__ spline_w,
    _Float16* __restrict__ W)
{
    int tid = blockIdx.x * blockDim.x + threadIdx.x;   // over CH*CH (o,i)
    if (tid >= CH * CH) return;
    const GLOBAL(float)* sp = (const GLOBAL(float)*)(spline_w + (size_t)tid * KC);
    const GLOBAL(float)* bw = (const GLOBAL(float)*)(base_w + tid);
    GLOBAL(_Float16)* w = (GLOBAL(_Float16)*)(W + (size_t)tid * KAUG); // (o*CH+i)*9
#pragma unroll
    for (int k = 0; k < KC; ++k) w[k] = (_Float16)sp[k];
    w[KC] = (_Float16)bw[0];
}

// ---------------------------------------------------------------------------
// Cox-de Boor cubic B-spline bases on the fixed uniform extended knot grid
// t[j] = -0.2 + (j-3)*0.08, j = 0..11. Writes 8 bases + x per element (f16).
// ---------------------------------------------------------------------------
__global__ __launch_bounds__(256) void expand_bases(
    const float* __restrict__ X,
    _Float16* __restrict__ A)
{
    int tid = blockIdx.x * blockDim.x + threadIdx.x;   // over N_TOK*CH (n,i)
    if (tid >= N_TOK * CH) return;
    const float x = ((const GLOBAL(float)*)X)[tid];

    float b[11];
#pragma unroll
    for (int j = 0; j < 11; ++j) {
        const float t0 = -0.2f + (float)(j - 3) * 0.08f;
        const float t1 = -0.2f + (float)(j - 2) * 0.08f;
        b[j] = (x >= t0 && x < t1) ? 1.0f : 0.0f;
    }
#pragma unroll
    for (int k = 1; k <= 3; ++k) {
        const float inv = 1.0f / ((float)k * 0.08f);
#pragma unroll
        for (int j = 0; j < 11; ++j) {
            if (j < 11 - k) {
                const float tj   = -0.2f + (float)(j - 3) * 0.08f;      // t[j]
                const float tjk1 = -0.2f + (float)(j + k - 2) * 0.08f;  // t[j+k+1]
                b[j] = (x - tj) * inv * b[j] + (tjk1 - x) * inv * b[j + 1];
            }
        }
    }

    GLOBAL(_Float16)* a = (GLOBAL(_Float16)*)(A + (size_t)tid * KAUG); // n*KDIM + i*9
#pragma unroll
    for (int k = 0; k < KC; ++k) a[k] = (_Float16)b[k];
    a[KC] = (_Float16)x;                    // base (identity-activation) term
}

// ---------------------------------------------------------------------------
// WMMA GEMM:  C[M,N] = A[M,K](f16, row-major) * W[N,K](f16, row-major)^T
// Block tile 128x128, 4 waves (2x2), each wave 64x64 = 4x4 WMMA accumulators.
// A fragment (16x32, M x K): lane = M row (lane&15); 16 contiguous K values at
//   K = (lane>>4)*16 (two 16B chunks per lane).
// B fragment (32x16, K x N): lane = N col (lane&15); 16 contiguous K values at
//   K = (lane>>4)*16 (two 16B chunks per lane).  [own K order, same for A and B]
// ---------------------------------------------------------------------------
__device__ __forceinline__ half16 cat16(half8 lo, half8 hi)
{
    return __builtin_shufflevector(lo, hi, 0, 1, 2, 3, 4, 5, 6, 7,
                                           8, 9, 10, 11, 12, 13, 14, 15);
}

__global__ __launch_bounds__(128) void kan_wmma_gemm(
    const _Float16* __restrict__ A,
    const _Float16* __restrict__ W,
    float* __restrict__ C,
    int M, int N, int K)
{
    const int lane  = threadIdx.x & 31;
    const int wave  = threadIdx.x >> 5;   // 0..3
    const int waveM = wave >> 1;          // 0..1
    const int waveN = wave & 1;           // 0..1
    const int m0 = blockIdx.x * 128 + waveM * 64;
    const int n0 = blockIdx.y * 128 + waveN * 64;
    const int lr = lane & 15;             // row-within-tile for A, col for B
    const int lh = lane >> 4;             // half-wave selector

    const GLOBAL(_Float16)* Ag = (const GLOBAL(_Float16)*)A;
    const GLOBAL(_Float16)* Wg = (const GLOBAL(_Float16)*)W;
    GLOBAL(float)* Cg = (GLOBAL(float)*)C;

    const GLOBAL(_Float16)* ap[4];
    const GLOBAL(_Float16)* bp[4];
#pragma unroll
    for (int t = 0; t < 4; ++t) {
        ap[t] = Ag + (size_t)(m0 + t * 16 + lr) * K + lh * 16;
        bp[t] = Wg + (size_t)(n0 + t * 16 + lr) * K + lh * 16;
    }

    const floatx8 zero = {0.f, 0.f, 0.f, 0.f, 0.f, 0.f, 0.f, 0.f};
    floatx8 acc[4][4];
#pragma unroll
    for (int mt = 0; mt < 4; ++mt)
#pragma unroll
        for (int nt = 0; nt < 4; ++nt) acc[mt][nt] = zero;

    for (int k0 = 0; k0 < K; k0 += 32) {
        half16 af[4], bf[4];
#pragma unroll
        for (int t = 0; t < 4; ++t) {
            const half8 alo = *(const GLOBAL(half8)*)(ap[t]);
            const half8 ahi = *(const GLOBAL(half8)*)(ap[t] + 8);
            af[t] = cat16(alo, ahi);
            const half8 blo = *(const GLOBAL(half8)*)(bp[t]);
            const half8 bhi = *(const GLOBAL(half8)*)(bp[t] + 8);
            bf[t] = cat16(blo, bhi);
            // keep-near prefetch of the stream four iterations ahead
            __builtin_prefetch((const GLOBAL(char)*)ap[t] + 256, 0, 3);
            __builtin_prefetch((const GLOBAL(char)*)bp[t] + 256, 0, 3);
            ap[t] += 32;
            bp[t] += 32;
        }
#pragma unroll
        for (int mt = 0; mt < 4; ++mt)
#pragma unroll
            for (int nt = 0; nt < 4; ++nt)
                acc[mt][nt] = __builtin_amdgcn_wmma_f32_16x16x32_f16(
                    /*neg_a=*/false, af[mt], /*neg_b=*/false, bf[nt],
                    /*c_mod=*/(short)0, acc[mt][nt],
                    /*reuse_a=*/false, /*reuse_b=*/false);
    }

    // C/D layout: VGPR r -> M = r + (lane>>4)*8 ; N = lane&15
#pragma unroll
    for (int mt = 0; mt < 4; ++mt) {
#pragma unroll
        for (int nt = 0; nt < 4; ++nt) {
#pragma unroll
            for (int r = 0; r < 8; ++r) {
                const int m = m0 + mt * 16 + r + lh * 8;
                const int n = n0 + nt * 16 + lr;
                Cg[(size_t)m * N + n] = acc[mt][nt][r];
            }
        }
    }
}

// ---------------------------------------------------------------------------
// Launch: pack W1, expand x, GEMM -> X1 ; pack W2, expand X1, GEMM -> out
// Workspace layout: [A_aug f16 144MB][W_aug f16 18MB][X1 f32 32MB] (~195MB)
// ---------------------------------------------------------------------------
extern "C" void kernel_launch(void* const* d_in, const int* in_sizes, int n_in,
                              void* d_out, int out_size, void* d_ws, size_t ws_size,
                              hipStream_t stream)
{
    const float* x         = (const float*)d_in[0];
    const float* base_w1   = (const float*)d_in[1];
    const float* spline_w1 = (const float*)d_in[2];
    const float* base_w2   = (const float*)d_in[3];
    const float* spline_w2 = (const float*)d_in[4];
    float* out = (float*)d_out;

    char* ws = (char*)d_ws;
    _Float16* Aaug = (_Float16*)ws;
    _Float16* Wp   = (_Float16*)(ws + A_BYTES);
    float*    X1   = (float*)(ws + A_BYTES + W_BYTES);

    const dim3 bS(256);
    const dim3 gW((CH * CH + 255) / 256);
    const dim3 gE((N_TOK * CH + 255) / 256);
    const dim3 bG(128);
    const dim3 gG(N_TOK / 128, CH / 128);

    // Layer 1
    pack_weights<<<gW, bS, 0, stream>>>(base_w1, spline_w1, Wp);
    expand_bases<<<gE, bS, 0, stream>>>(x, Aaug);
    kan_wmma_gemm<<<gG, bG, 0, stream>>>(Aaug, Wp, X1, N_TOK, CH, KDIM);

    // Layer 2
    pack_weights<<<gW, bS, 0, stream>>>(base_w2, spline_w2, Wp);
    expand_bases<<<gE, bS, 0, stream>>>(X1, Aaug);
    kan_wmma_gemm<<<gG, bG, 0, stream>>>(Aaug, Wp, out, N_TOK, CH, KDIM);
}
